// MultiHeadAttn_86431921865309
// MI455X (gfx1250) — matrix operand order repaired and checked by reading, not compiled, not run on hardware
//
#include <hip/hip_runtime.h>
#include <hip/hip_bf16.h>

typedef __attribute__((ext_vector_type(16))) _Float16 v16h;
typedef __attribute__((ext_vector_type(8)))  float    v8f;
typedef __attribute__((ext_vector_type(4)))  int      v4i;

#define D_   1024
#define H_   8
#define B_   8
#define S_   1024
#define DH_  128
#define LDSP 40   // padded LDS row stride in halves (32 data + 8 pad)

// gfx1250 async global->LDS copy path (ASYNCcnt-tracked, bypasses VGPRs).
#if defined(__has_builtin)
#if __has_builtin(__builtin_amdgcn_global_load_async_to_lds_b128) && \
    __has_builtin(__builtin_amdgcn_s_wait_asynccnt)
#define USE_ASYNC_LDS 1
#endif
#endif
#ifndef USE_ASYNC_LDS
#define USE_ASYNC_LDS 0
#endif

#define AS1 __attribute__((address_space(1)))
#define AS3 __attribute__((address_space(3)))

// ---------------------------------------------------------------------------
// WMMA fragment loaders (wave32, 16x16x32 f16 -> f32)
// A is MxK (16x32): lane holds row M=lane%16; lanes<16 hold K{0..7,16..23},
// lanes>=16 hold K{8..15,24..31} (ISA 7.12.2 16-bit A layout).
// B is KxN (32x16): lane holds col N=lane%16; lanes<16 hold K{0..7,16..23},
// lanes>=16 hold K{8..15,24..31} (ISA 7.12.4 B layout pattern).
// LDS tiles store both operands as T[n_or_m][k] with row stride LDSP, so the
// same contiguous-along-k read pattern serves A and B in every kernel.
// ---------------------------------------------------------------------------
__device__ __forceinline__ v16h frag_a(const _Float16* As, int wv, int lane) {
  int lmod = lane & 15, lh = lane >> 4;
  const _Float16* p = As + (16 * wv + lmod) * LDSP + lh * 8;
  v16h a;
#pragma unroll
  for (int i = 0; i < 8; ++i) { a[i] = p[i]; a[8 + i] = p[16 + i]; }
  return a;
}

__device__ __forceinline__ v16h frag_b(const _Float16* Bs, int j, int lane) {
  int lmod = lane & 15, lh = lane >> 4;
  const _Float16* p = Bs + (16 * j + lmod) * LDSP + lh * 8;
  v16h b;
#pragma unroll
  for (int i = 0; i < 8; ++i) { b[i] = p[i]; b[8 + i] = p[16 + i]; }
  return b;
}

// ---------------------------------------------------------------------------
// K0: qmask[b*S+q] = sign(|sum_d queries[b,q,d]|)  (one wave per row)
// ---------------------------------------------------------------------------
__global__ void mha_qmask_kernel(const float* __restrict__ q,
                                 float* __restrict__ qmask) {
  int row  = blockIdx.x * 8 + (threadIdx.x >> 5);
  int lane = threadIdx.x & 31;
  const float* p = q + (size_t)row * D_;
  float s = 0.0f;
  for (int i = lane; i < D_; i += 32) s += p[i];
#pragma unroll
  for (int off = 16; off > 0; off >>= 1) s += __shfl_down(s, off, 32);
  if (lane == 0) qmask[row] = (s != 0.0f) ? 1.0f : 0.0f;
}

// ---------------------------------------------------------------------------
// K1: P = relu(X @ W^T + bias) in f16.  X:(8192,1024) f32, W:(1024,1024) f32.
// 128x128 tile per block, 8 waves, wave w owns rows [16w,16w+16).
// Staging converts f32->f16 in flight (so no async copy); next K-chunk is
// prefetched with global_prefetch_b8.
// ---------------------------------------------------------------------------
__global__ void mha_proj_kernel(const float* __restrict__ X,
                                const float* __restrict__ W,
                                const float* __restrict__ bias,
                                _Float16* __restrict__ P) {
  __shared__ __align__(16) _Float16 As[128 * LDSP];
  __shared__ __align__(16) _Float16 Bs[128 * LDSP];
  int m0 = blockIdx.x * 128;
  int n0 = blockIdx.y * 128;
  int t = threadIdx.x, lane = t & 31, wv = t >> 5;
  int r = t >> 1, hseg = t & 1;                 // staging: 2 threads per row

  v8f acc[8];
#pragma unroll
  for (int j = 0; j < 8; ++j)
#pragma unroll
    for (int i = 0; i < 8; ++i) acc[j][i] = 0.0f;

  const float* xrow = X + (size_t)(m0 + r) * D_ + hseg * 16;
  const float* wrow = W + (size_t)(n0 + r) * D_ + hseg * 16;

  for (int k0 = 0; k0 < D_; k0 += 32) {
    __syncthreads();
    if (k0 + 32 < D_) {                      // prefetch next chunk into caches
      __builtin_prefetch(xrow + k0 + 32, 0, 0);
      __builtin_prefetch(wrow + k0 + 32, 0, 0);
    }
    {
      const float4* src = (const float4*)(xrow + k0);
      _Float16* dst = &As[r * LDSP + hseg * 16];
#pragma unroll
      for (int j = 0; j < 4; ++j) {
        float4 v = src[j];
        dst[4 * j + 0] = (_Float16)v.x; dst[4 * j + 1] = (_Float16)v.y;
        dst[4 * j + 2] = (_Float16)v.z; dst[4 * j + 3] = (_Float16)v.w;
      }
    }
    {
      const float4* src = (const float4*)(wrow + k0);
      _Float16* dst = &Bs[r * LDSP + hseg * 16];
#pragma unroll
      for (int j = 0; j < 4; ++j) {
        float4 v = src[j];
        dst[4 * j + 0] = (_Float16)v.x; dst[4 * j + 1] = (_Float16)v.y;
        dst[4 * j + 2] = (_Float16)v.z; dst[4 * j + 3] = (_Float16)v.w;
      }
    }
    __syncthreads();
    v16h a = frag_a(As, wv, lane);
    v16h bcur = frag_b(Bs, 0, lane);
#pragma unroll
    for (int j = 0; j < 8; ++j) {            // pipeline: load B[j+1] before WMMA[j]
      v16h bnext = frag_b(Bs, (j + 1) & 7, lane);
      acc[j] = __builtin_amdgcn_wmma_f32_16x16x32_f16(false, a, false, bcur,
                                                      (short)0, acc[j], false, false);
      bcur = bnext;
    }
  }

  int lmod = lane & 15, lh = lane >> 4;
#pragma unroll
  for (int j = 0; j < 8; ++j) {
    int col = n0 + 16 * j + lmod;
    float bv = bias[col];
#pragma unroll
    for (int v = 0; v < 8; ++v) {
      int row = m0 + 16 * wv + v + 8 * lh;
      float x = fmaxf(acc[j][v] + bv, 0.0f);
      P[(size_t)row * D_ + col] = (_Float16)x;
    }
  }
}

// ---------------------------------------------------------------------------
// K2: attn_score[z, q, k] = (Qh . Kh) * scale * qmask   (z = h*B + b)
// Q/K f16 stored as (b,s,D) with head offset h*DH; reduction dim DH=128.
// Tiles are pure f16 -> staged with gfx1250 async global->LDS copies.
// ---------------------------------------------------------------------------
__global__ void mha_scores_kernel(const _Float16* __restrict__ Q,
                                  const _Float16* __restrict__ K,
                                  const float* __restrict__ qmask,
                                  float* __restrict__ attn_score) {
  __shared__ __align__(16) _Float16 As[128 * LDSP];
  __shared__ __align__(16) _Float16 Bs[128 * LDSP];
  int kt0 = blockIdx.x * 128;
  int q0  = blockIdx.y * 128;
  int z   = blockIdx.z;
  int h = z >> 3, b = z & 7;
  int t = threadIdx.x, lane = t & 31, wv = t >> 5;
  int r = t >> 1, hseg = t & 1;

  v8f acc[8];
#pragma unroll
  for (int j = 0; j < 8; ++j)
#pragma unroll
    for (int i = 0; i < 8; ++i) acc[j][i] = 0.0f;

  const size_t hb = (size_t)b * S_ * D_ + (size_t)h * DH_;
  const _Float16* qsrc = Q + hb + (size_t)(q0 + r) * D_ + hseg * 16;
  const _Float16* ksrc = K + hb + (size_t)(kt0 + r) * D_ + hseg * 16;
  _Float16* adst = &As[r * LDSP + hseg * 16];
  _Float16* bdst = &Bs[r * LDSP + hseg * 16];

  for (int k0 = 0; k0 < DH_; k0 += 32) {
    __syncthreads();
#if USE_ASYNC_LDS
    __builtin_amdgcn_global_load_async_to_lds_b128(
        (AS1 v4i*)(qsrc + k0), (AS3 v4i*)adst, 0, 0);
    __builtin_amdgcn_global_load_async_to_lds_b128(
        (AS1 v4i*)(qsrc + k0), (AS3 v4i*)adst, 16, 0);
    __builtin_amdgcn_global_load_async_to_lds_b128(
        (AS1 v4i*)(ksrc + k0), (AS3 v4i*)bdst, 0, 0);
    __builtin_amdgcn_global_load_async_to_lds_b128(
        (AS1 v4i*)(ksrc + k0), (AS3 v4i*)bdst, 16, 0);
    __builtin_amdgcn_s_wait_asynccnt(0);
#else
    {
      const uint4* src = (const uint4*)(qsrc + k0);
      uint4* dst = (uint4*)adst;
      dst[0] = src[0]; dst[1] = src[1];
    }
    {
      const uint4* src = (const uint4*)(ksrc + k0);
      uint4* dst = (uint4*)bdst;
      dst[0] = src[0]; dst[1] = src[1];
    }
#endif
    __syncthreads();
    v16h a = frag_a(As, wv, lane);
    v16h bcur = frag_b(Bs, 0, lane);
#pragma unroll
    for (int j = 0; j < 8; ++j) {
      v16h bnext = frag_b(Bs, (j + 1) & 7, lane);
      acc[j] = __builtin_amdgcn_wmma_f32_16x16x32_f16(false, a, false, bcur,
                                                      (short)0, acc[j], false, false);
      bcur = bnext;
    }
  }

  const float scale = 0.08838834764831845f;   // 1/sqrt(128)
  int lmod = lane & 15, lh = lane >> 4;
#pragma unroll
  for (int v = 0; v < 8; ++v) {
    int row = q0 + 16 * wv + v + 8 * lh;
    float mk = qmask[b * S_ + row] * scale;
#pragma unroll
    for (int j = 0; j < 8; ++j) {
      int col = kt0 + 16 * j + lmod;
      attn_score[((size_t)z * S_ + row) * S_ + col] = acc[j][v] * mk;
    }
  }
}

// ---------------------------------------------------------------------------
// K3: per-row softmax stats over stored scores (wave per row)
// ---------------------------------------------------------------------------
__global__ void mha_stats_kernel(const float* __restrict__ sc,
                                 float* __restrict__ rmax,
                                 float* __restrict__ rsum) {
  int rid  = blockIdx.x * 8 + (threadIdx.x >> 5);
  int lane = threadIdx.x & 31;
  const float* p = sc + (size_t)rid * S_;
  float m = -3.4e38f;
  for (int i = lane; i < S_; i += 32) m = fmaxf(m, p[i]);
#pragma unroll
  for (int off = 16; off > 0; off >>= 1) m = fmaxf(m, __shfl_down(m, off, 32));
  m = __shfl(m, 0, 32);
  float s = 0.0f;
  for (int i = lane; i < S_; i += 32) s += __expf(p[i] - m);
#pragma unroll
  for (int off = 16; off > 0; off >>= 1) s += __shfl_down(s, off, 32);
  if (lane == 0) { rmax[rid] = m; rsum[rid] = s; }
}

// ---------------------------------------------------------------------------
// K4: out[b,q,h*DH+d] = qmask/L * sum_k exp(score-m) * V[k,d]
// A = exp(score-m) f16 (128x32 chunk), B = V staged transposed in LDS.
// ---------------------------------------------------------------------------
__global__ void mha_out_kernel(const float* __restrict__ sc,
                               const _Float16* __restrict__ V,
                               const float* __restrict__ rmax,
                               const float* __restrict__ rsum,
                               const float* __restrict__ qmask,
                               float* __restrict__ out) {
  __shared__ __align__(16) _Float16 As[128 * LDSP];
  __shared__ __align__(16) _Float16 Bs[128 * LDSP];
  int q0 = blockIdx.x * 128;
  int z  = blockIdx.y;
  int h = z >> 3, b = z & 7;
  int t = threadIdx.x, lane = t & 31, wv = t >> 5;
  int r = t >> 1, hseg = t & 1;          // A staging: 2 threads/row
  int krow = t >> 3, seg = t & 7;        // B staging: 8 threads/V-row

  v8f acc[8];
#pragma unroll
  for (int j = 0; j < 8; ++j)
#pragma unroll
    for (int i = 0; i < 8; ++i) acc[j][i] = 0.0f;

  const float* scz  = sc + (size_t)z * S_ * S_ + (size_t)(q0 + r) * S_ + hseg * 16;
  const float  mrow = rmax[z * S_ + q0 + r];

  for (int kc = 0; kc < S_; kc += 32) {
    __syncthreads();
    if (kc + 32 < S_) __builtin_prefetch(scz + kc + 32, 0, 0);
    {   // A: attn numerator in f16
      const float4* src = (const float4*)(scz + kc);
      _Float16* dst = &As[r * LDSP + hseg * 16];
#pragma unroll
      for (int j = 0; j < 4; ++j) {
        float4 v4 = src[j];
        dst[4 * j + 0] = (_Float16)__expf(v4.x - mrow);
        dst[4 * j + 1] = (_Float16)__expf(v4.y - mrow);
        dst[4 * j + 2] = (_Float16)__expf(v4.z - mrow);
        dst[4 * j + 3] = (_Float16)__expf(v4.w - mrow);
      }
    }
    {   // B: V tile, transposed into Bs[dh][k]
      const _Float16* src = V + ((size_t)(b * S_ + kc + krow)) * D_ + h * DH_ + seg * 16;
      _Float16 tmp[16];
      *(uint4*)&tmp[0] = *(const uint4*)src;
      *(uint4*)&tmp[8] = *(const uint4*)(src + 8);
#pragma unroll
      for (int i = 0; i < 16; ++i) Bs[(seg * 16 + i) * LDSP + krow] = tmp[i];
    }
    __syncthreads();
    v16h a = frag_a(As, wv, lane);
    v16h bcur = frag_b(Bs, 0, lane);
#pragma unroll
    for (int j = 0; j < 8; ++j) {
      v16h bnext = frag_b(Bs, (j + 1) & 7, lane);
      acc[j] = __builtin_amdgcn_wmma_f32_16x16x32_f16(false, a, false, bcur,
                                                      (short)0, acc[j], false, false);
      bcur = bnext;
    }
  }

  int lmod = lane & 15, lh = lane >> 4;
#pragma unroll
  for (int v = 0; v < 8; ++v) {
    int row = q0 + 16 * wv + v + 8 * lh;
    float w = qmask[b * S_ + row] / rsum[z * S_ + row];
#pragma unroll
    for (int j = 0; j < 8; ++j) {
      int col = 16 * j + lmod;
      out[((size_t)b * S_ + row) * D_ + h * DH_ + col] = acc[j][v] * w;
    }
  }
}

// ---------------------------------------------------------------------------
extern "C" void kernel_launch(void* const* d_in, const int* in_sizes, int n_in,
                              void* d_out, int out_size, void* d_ws, size_t ws_size,
                              hipStream_t stream) {
  (void)in_sizes; (void)n_in; (void)out_size; (void)ws_size;
  const float* queries = (const float*)d_in[0];
  const float* keys    = (const float*)d_in[1];
  const float* values  = (const float*)d_in[2];
  const float* Wq = (const float*)d_in[3];
  const float* bq = (const float*)d_in[4];
  const float* Wk = (const float*)d_in[5];
  const float* bk = (const float*)d_in[6];
  const float* Wv = (const float*)d_in[7];
  const float* bv = (const float*)d_in[8];

  float* out        = (float*)d_out;                    // (B,S,D)
  float* attn_score = out + (size_t)B_ * S_ * D_;       // (H*B,S,S)

  const size_t PE = (size_t)B_ * S_ * D_;               // 8388608 elements
  _Float16* Qh = (_Float16*)d_ws;
  _Float16* Kh = Qh + PE;
  _Float16* Vh = Kh + PE;
  float* qmask = (float*)(Vh + PE);
  float* rmax  = qmask + (size_t)B_ * S_;
  float* rsum  = rmax + (size_t)H_ * B_ * S_;

  mha_qmask_kernel <<<(B_ * S_) / 8, 256, 0, stream>>>(queries, qmask);
  mha_proj_kernel  <<<dim3(64, 8), 256, 0, stream>>>(queries, Wq, bq, Qh);
  mha_proj_kernel  <<<dim3(64, 8), 256, 0, stream>>>(keys,    Wk, bk, Kh);
  mha_proj_kernel  <<<dim3(64, 8), 256, 0, stream>>>(values,  Wv, bv, Vh);
  mha_scores_kernel<<<dim3(8, 8, 64), 256, 0, stream>>>(Qh, Kh, qmask, attn_score);
  mha_stats_kernel <<<(H_ * B_ * S_) / 8, 256, 0, stream>>>(attn_score, rmax, rsum);
  mha_out_kernel   <<<dim3(8, 64), 256, 0, stream>>>(attn_score, Vh, rmax, rsum, qmask, out);
}
